// Decoder_29626684408461
// MI455X (gfx1250) — matrix operand order repaired and checked by reading, not compiled, not run on hardware
//
#include <hip/hip_runtime.h>

// ---------------------------------------------------------------------------
// Tacotron-style attention decoder for MI455X (gfx1250, wave32, WMMA).
// All GEMMs use v_wmma_f32_16x16x32_bf16 (bf16 in, f32 accumulate).
// Hot tensors (pk, encoder, weights) are converted to bf16 once so the whole
// per-step working set (~137 MB) is L2-resident (192 MB L2, 23.3 TB/s HBM).
// Workspace requirement: ~141 MB.
// ---------------------------------------------------------------------------

typedef __bf16 bf16_t;
typedef __attribute__((ext_vector_type(16))) __bf16 v16bf;
typedef __attribute__((ext_vector_type(8)))  __bf16 v8bf;
typedef __attribute__((ext_vector_type(8)))  float  v8f;

#define B_     64
#define SENC_  512
#define T_     800
#define ENC_   512
#define DEC_   1024
#define MEL_   80
#define G4_    4096          // 4*DEC
#define KA_    1632          // [mel 80 | enc_last 512 | pad 16 | h 1024]
#define KAH_   608           // h offset inside XA
#define KD_    2560          // [ah 1024 | context 512 | dh 1024]
#define KDH_   1536          // dh offset inside XD

__device__ __forceinline__ float sigm(float x) { return 1.f / (1.f + __expf(-x)); }

// A-operand tile: 16x32 bf16. Lane L: m = m0+(L&15); lane half h holds k0+16h..k0+16h+15 (contiguous, as in B).
__device__ __forceinline__ v16bf load_A16x32(const bf16_t* __restrict__ X, int ld,
                                             int m0, int k0) {
    int lane = threadIdx.x & 31;
    int m  = m0 + (lane & 15);
    int kb = (lane & 16) ? 16 : 0;
    const bf16_t* p = X + (size_t)m * ld + k0 + kb;
    v8bf lo = *(const v8bf*)(p);
    v8bf hi = *(const v8bf*)(p + 8);
    return __builtin_shufflevector(lo, hi, 0,1,2,3,4,5,6,7,8,9,10,11,12,13,14,15);
}

// B-operand tile: 32x16 bf16 with B[k][n] = W[n][k]; W row-major (n,k) so each
// lane's 16 k-values are contiguous (no transpose needed).
__device__ __forceinline__ v16bf load_B32x16(const bf16_t* __restrict__ W, int ld,
                                             int n0, int k0) {
    int lane = threadIdx.x & 31;
    int n  = n0 + (lane & 15);
    int kb = (lane & 16) ? 16 : 0;
    const bf16_t* p = W + (size_t)n * ld + k0 + kb;
    return *(const v16bf*)p;
}

#define WMMA_BF16(a, b, c) \
    __builtin_amdgcn_wmma_f32_16x16x32_bf16(false, (a), false, (b), (short)0, (c), false, false)

// ---------------------------------------------------------------------------
// Fused LSTM: gates = X @ Wcat^T (+bias), cell update, h written as bf16 into
// (up to) two destination X buffers. Block = 128 thr (4 waves = 4 M-tiles of
// B=64). Grid.x = 64 (unit columns / 16). Each wave: 4 WMMAs (i,f,g,o) per
// 32-wide K step.
// ---------------------------------------------------------------------------
__global__ void lstm_gates_kernel(const bf16_t* __restrict__ X, int ldx, int K,
                                  const bf16_t* __restrict__ W,   // [4096][K]
                                  const float*  __restrict__ bias,// [4096]
                                  float* __restrict__ cstate,     // [64][1024]
                                  bf16_t* __restrict__ H1, int ldh1, int off1,
                                  bf16_t* __restrict__ H2, int ldh2, int off2) {
    int wave = threadIdx.x >> 5;
    int lane = threadIdx.x & 31;
    int m0 = wave * 16;
    int u0 = blockIdx.x * 16;
    v8f acc0 = {}, acc1 = {}, acc2 = {}, acc3 = {};
    for (int k0 = 0; k0 < K; k0 += 32) {
        v16bf a = load_A16x32(X, ldx, m0, k0);
        v16bf b0 = load_B32x16(W, K, u0,            k0);
        acc0 = WMMA_BF16(a, b0, acc0);
        v16bf b1 = load_B32x16(W, K, 1024 + u0,     k0);
        acc1 = WMMA_BF16(a, b1, acc1);
        v16bf b2 = load_B32x16(W, K, 2048 + u0,     k0);
        acc2 = WMMA_BF16(a, b2, acc2);
        v16bf b3 = load_B32x16(W, K, 3072 + u0,     k0);
        acc3 = WMMA_BF16(a, b3, acc3);
    }
    int u = u0 + (lane & 15);
    float bi = bias[u], bff = bias[1024 + u], bg = bias[2048 + u], bo = bias[3072 + u];
#pragma unroll
    for (int v = 0; v < 8; ++v) {
        int m = m0 + v + ((lane & 16) ? 8 : 0);
        float gi = sigm(acc0[v] + bi);
        float gf = sigm(acc1[v] + bff);
        float gg = tanhf(acc2[v] + bg);
        float go = sigm(acc3[v] + bo);
        size_t ci = (size_t)m * DEC_ + u;
        float cn = gf * cstate[ci] + gi * gg;
        cstate[ci] = cn;
        bf16_t h = (bf16_t)(go * tanhf(cn));
        H1[(size_t)m * ldh1 + off1 + u] = h;
        if (H2) H2[(size_t)m * ldh2 + off2 + u] = h;
    }
}

// Generic bf16 GEMM, f32 out.  grid(x=N/16, y=Mtiles/4), block=128.
__global__ void gemm_f32out(const bf16_t* __restrict__ A, int lda,
                            const bf16_t* __restrict__ Bw, int ldb,
                            float* __restrict__ C, int ldc, int K) {
    int wave = threadIdx.x >> 5, lane = threadIdx.x & 31;
    int m0 = (blockIdx.y * 4 + wave) * 16;
    int n0 = blockIdx.x * 16;
    v8f acc = {};
    for (int k0 = 0; k0 < K; k0 += 32) {
        v16bf a = load_A16x32(A, lda, m0, k0);
        v16bf b = load_B32x16(Bw, ldb, n0, k0);
        acc = WMMA_BF16(a, b, acc);
    }
    int n = n0 + (lane & 15);
#pragma unroll
    for (int v = 0; v < 8; ++v) {
        int m = m0 + v + ((lane & 16) ? 8 : 0);
        C[(size_t)m * ldc + n] = acc[v];
    }
}

// Same, bf16 out (used for pk so the 67 MB tensor stays L2-resident).
__global__ void gemm_bf16out(const bf16_t* __restrict__ A, int lda,
                             const bf16_t* __restrict__ Bw, int ldb,
                             bf16_t* __restrict__ C, int ldc, int K) {
    int wave = threadIdx.x >> 5, lane = threadIdx.x & 31;
    int m0 = (blockIdx.y * 4 + wave) * 16;
    int n0 = blockIdx.x * 16;
    v8f acc = {};
    for (int k0 = 0; k0 < K; k0 += 32) {
        v16bf a = load_A16x32(A, lda, m0, k0);
        v16bf b = load_B32x16(Bw, ldb, n0, k0);
        acc = WMMA_BF16(a, b, acc);
    }
    int n = n0 + (lane & 15);
#pragma unroll
    for (int v = 0; v < 8; ++v) {
        int m = m0 + v + ((lane & 16) ? 8 : 0);
        C[(size_t)m * ldc + n] = (bf16_t)acc[v];
    }
}

// Mel projection: M=64, N=80 (5 tiles), K=1024, + bias, scattered into d_out.
__global__ void mel_gemm_kernel(const bf16_t* __restrict__ A, int lda,
                                const bf16_t* __restrict__ Wm,  // [80][1024]
                                const float* __restrict__ bmel,
                                float* __restrict__ out, int t) {
    int wave = threadIdx.x >> 5, lane = threadIdx.x & 31;
    int m0 = wave * 16;
    int n0 = blockIdx.x * 16;
    v8f acc = {};
    for (int k0 = 0; k0 < DEC_; k0 += 32) {
        v16bf a = load_A16x32(A, lda, m0, k0);
        v16bf b = load_B32x16(Wm, DEC_, n0, k0);
        acc = WMMA_BF16(a, b, acc);
    }
    int n = n0 + (lane & 15);
    float bn = bmel[n];
#pragma unroll
    for (int v = 0; v < 8; ++v) {
        int m = m0 + v + ((lane & 16) ? 8 : 0);
        out[((size_t)m * T_ + t) * MEL_ + n] = acc[v] + bn;
    }
}

// scores[b][s] = sum_d We[d]*tanh(pq[b][d] + pk[b][s][d]).  grid(64,64), 256thr.
__global__ void attn_scores_kernel(const float* __restrict__ pq,
                                   const bf16_t* __restrict__ pk,
                                   const float* __restrict__ We,
                                   float* __restrict__ scores) {
    int b = blockIdx.x;
    int wave = threadIdx.x >> 5, lane = threadIdx.x & 31;
    int s = blockIdx.y * 8 + wave;
    const float*  pqb = pq + (size_t)b * DEC_;
    const bf16_t* pkp = pk + ((size_t)b * SENC_ + s) * DEC_;
    float acc = 0.f;
    for (int d0 = lane * 8; d0 < DEC_; d0 += 32 * 8) {
        v8bf pv = *(const v8bf*)(pkp + d0);
#pragma unroll
        for (int j = 0; j < 8; ++j)
            acc += We[d0 + j] * tanhf(pqb[d0 + j] + (float)pv[j]);
    }
#pragma unroll
    for (int o = 16; o; o >>= 1) acc += __shfl_xor(acc, o, 32);
    if (lane == 0) scores[(size_t)b * SENC_ + s] = acc;
}

// softmax over S=512 + context matvec; writes bf16 context into XD[:,1024:1536].
__global__ void softmax_ctx_kernel(const float* __restrict__ scores,
                                   const bf16_t* __restrict__ encb,
                                   bf16_t* __restrict__ XD) {
    __shared__ float w[SENC_];
    __shared__ float red[256];
    int b = blockIdx.x, tid = threadIdx.x;
    float s0 = scores[(size_t)b * SENC_ + tid];
    float s1 = scores[(size_t)b * SENC_ + 256 + tid];
    red[tid] = fmaxf(s0, s1);
    __syncthreads();
    for (int o = 128; o; o >>= 1) { if (tid < o) red[tid] = fmaxf(red[tid], red[tid + o]); __syncthreads(); }
    float mx = red[0];
    __syncthreads();
    float e0 = __expf(s0 - mx), e1 = __expf(s1 - mx);
    w[tid] = e0; w[tid + 256] = e1;
    red[tid] = e0 + e1;
    __syncthreads();
    for (int o = 128; o; o >>= 1) { if (tid < o) red[tid] += red[tid + o]; __syncthreads(); }
    float inv = 1.f / red[0];
    float c0 = 0.f, c1 = 0.f;
    const bf16_t* eb = encb + (size_t)b * SENC_ * ENC_;
    for (int s = 0; s < SENC_; ++s) {
        float ws = w[s];
        c0 += ws * (float)eb[(size_t)s * ENC_ + tid];
        c1 += ws * (float)eb[(size_t)s * ENC_ + tid + 256];
    }
    XD[(size_t)b * KD_ + 1024 + tid]       = (bf16_t)(c0 * inv);
    XD[(size_t)b * KD_ + 1024 + tid + 256] = (bf16_t)(c1 * inv);
}

// ------------------------------ init kernels -------------------------------
__global__ void build_wcat_kernel(const float* __restrict__ Wih, int Kih,
                                  const float* __restrict__ Whh, int Khh,
                                  bf16_t* __restrict__ Wc, int Kc, int hoff, long n) {
    long i = (long)blockIdx.x * blockDim.x + threadIdx.x;
    if (i >= n) return;
    int r = (int)(i / Kc), cc = (int)(i % Kc);
    float v = 0.f;
    if (cc < Kih)       v = Wih[(size_t)r * Kih + cc];
    else if (cc >= hoff) v = Whh[(size_t)r * Khh + (cc - hoff)];
    Wc[i] = (bf16_t)v;
}

__global__ void cvt_f32_bf16_kernel(const float* __restrict__ s, bf16_t* __restrict__ d, long n) {
    long i = (long)blockIdx.x * blockDim.x + threadIdx.x;
    if (i < n) d[i] = (bf16_t)s[i];
}

__global__ void vec_add_kernel(const float* __restrict__ a, const float* __restrict__ b,
                               float* __restrict__ d, int n) {
    int i = blockIdx.x * blockDim.x + threadIdx.x;
    if (i < n) d[i] = a[i] + b[i];
}

__global__ void init_XA_kernel(bf16_t* __restrict__ XA, const float* __restrict__ enc, long n) {
    long i = (long)blockIdx.x * blockDim.x + threadIdx.x;
    if (i >= n) return;
    int b = (int)(i / KA_), cc = (int)(i % KA_);
    float v = 0.f;
    if (cc >= 80 && cc < 592)
        v = enc[((size_t)b * SENC_ + (SENC_ - 1)) * ENC_ + (cc - 80)];
    XA[i] = (bf16_t)v;
}

__global__ void zero_bf16_kernel(bf16_t* __restrict__ d, long n) {
    long i = (long)blockIdx.x * blockDim.x + threadIdx.x;
    if (i < n) d[i] = (bf16_t)0.f;
}
__global__ void zero_f32_kernel(float* __restrict__ d, long n) {
    long i = (long)blockIdx.x * blockDim.x + threadIdx.x;
    if (i < n) d[i] = 0.f;
}

// Teacher-forced previous mel into XA[:, 0:80].  grid(64), block(80).
__global__ void set_mel_kernel(bf16_t* __restrict__ XA, const float* __restrict__ tm, int t) {
    int b = blockIdx.x, j = threadIdx.x;
    float v = (t > 0) ? tm[((size_t)b * T_ + (t - 1)) * MEL_ + j] : 0.f;
    XA[(size_t)b * KA_ + j] = (bf16_t)v;
}

// ---------------------------------------------------------------------------
extern "C" void kernel_launch(void* const* d_in, const int* in_sizes, int n_in,
                              void* d_out, int out_size, void* d_ws, size_t ws_size,
                              hipStream_t stream) {
    const float* enc   = (const float*)d_in[0];
    const float* tmel  = (const float*)d_in[1];
    const float* Wih_a = (const float*)d_in[2];
    const float* Whh_a = (const float*)d_in[3];
    const float* bih_a = (const float*)d_in[4];
    const float* bhh_a = (const float*)d_in[5];
    const float* Wq    = (const float*)d_in[6];
    const float* Wk    = (const float*)d_in[7];
    const float* We    = (const float*)d_in[8];
    const float* Wih_d = (const float*)d_in[9];
    const float* Whh_d = (const float*)d_in[10];
    const float* bih_d = (const float*)d_in[11];
    const float* bhh_d = (const float*)d_in[12];
    const float* Wmel  = (const float*)d_in[13];
    const float* bmel  = (const float*)d_in[14];
    float* out = (float*)d_out;

    char* ws = (char*)d_ws;
    size_t off = 0;
    auto carve = [&](size_t bytes) -> char* {
        char* p = ws + off;
        off += (bytes + 255) & ~(size_t)255;
        return p;
    };
    bf16_t* WcatA  = (bf16_t*)carve((size_t)G4_ * KA_ * 2);   // 13.4 MB
    bf16_t* WcatD  = (bf16_t*)carve((size_t)G4_ * KD_ * 2);   // 21.0 MB
    bf16_t* Wq_bf  = (bf16_t*)carve((size_t)DEC_ * DEC_ * 2); //  2.1 MB
    bf16_t* Wk_bf  = (bf16_t*)carve((size_t)DEC_ * ENC_ * 2); //  1.0 MB
    bf16_t* Wm_bf  = (bf16_t*)carve((size_t)MEL_ * DEC_ * 2); //  0.2 MB
    float*  b_a    = (float*)carve((size_t)G4_ * 4);
    float*  b_d    = (float*)carve((size_t)G4_ * 4);
    bf16_t* enc_bf = (bf16_t*)carve((size_t)B_ * SENC_ * ENC_ * 2);  // 33.6 MB
    bf16_t* pk     = (bf16_t*)carve((size_t)B_ * SENC_ * DEC_ * 2);  // 67.1 MB
    bf16_t* XA0    = (bf16_t*)carve((size_t)B_ * KA_ * 2);
    bf16_t* XA1    = (bf16_t*)carve((size_t)B_ * KA_ * 2);
    bf16_t* XD0    = (bf16_t*)carve((size_t)B_ * KD_ * 2);
    bf16_t* XD1    = (bf16_t*)carve((size_t)B_ * KD_ * 2);
    float*  ac     = (float*)carve((size_t)B_ * DEC_ * 4);
    float*  dc     = (float*)carve((size_t)B_ * DEC_ * 4);
    float*  pq     = (float*)carve((size_t)B_ * DEC_ * 4);
    float*  scores = (float*)carve((size_t)B_ * SENC_ * 4);

    const int thr = 256;
    auto nb = [&](long n) { return (unsigned)((n + thr - 1) / thr); };

    // ---- one-time (per-call) prep: bf16 weight/tensor staging ----
    build_wcat_kernel<<<nb((long)G4_ * KA_), thr, 0, stream>>>(Wih_a, 592, Whh_a, 1024, WcatA, KA_, KAH_, (long)G4_ * KA_);
    build_wcat_kernel<<<nb((long)G4_ * KD_), thr, 0, stream>>>(Wih_d, 1536, Whh_d, 1024, WcatD, KD_, KDH_, (long)G4_ * KD_);
    cvt_f32_bf16_kernel<<<nb((long)DEC_ * DEC_), thr, 0, stream>>>(Wq, Wq_bf, (long)DEC_ * DEC_);
    cvt_f32_bf16_kernel<<<nb((long)DEC_ * ENC_), thr, 0, stream>>>(Wk, Wk_bf, (long)DEC_ * ENC_);
    cvt_f32_bf16_kernel<<<nb((long)MEL_ * DEC_), thr, 0, stream>>>(Wmel, Wm_bf, (long)MEL_ * DEC_);
    cvt_f32_bf16_kernel<<<nb((long)B_ * SENC_ * ENC_), thr, 0, stream>>>(enc, enc_bf, (long)B_ * SENC_ * ENC_);
    vec_add_kernel<<<nb(G4_), thr, 0, stream>>>(bih_a, bhh_a, b_a, G4_);
    vec_add_kernel<<<nb(G4_), thr, 0, stream>>>(bih_d, bhh_d, b_d, G4_);
    init_XA_kernel<<<nb((long)B_ * KA_), thr, 0, stream>>>(XA0, enc, (long)B_ * KA_);
    init_XA_kernel<<<nb((long)B_ * KA_), thr, 0, stream>>>(XA1, enc, (long)B_ * KA_);
    zero_bf16_kernel<<<nb((long)B_ * KD_), thr, 0, stream>>>(XD0, (long)B_ * KD_);
    zero_bf16_kernel<<<nb((long)B_ * KD_), thr, 0, stream>>>(XD1, (long)B_ * KD_);
    zero_f32_kernel<<<nb((long)B_ * DEC_), thr, 0, stream>>>(ac, (long)B_ * DEC_);
    zero_f32_kernel<<<nb((long)B_ * DEC_), thr, 0, stream>>>(dc, (long)B_ * DEC_);

    // pk = enc @ Wk^T  (M=32768, N=1024, K=512) -> bf16, hoisted out of loop
    gemm_bf16out<<<dim3(DEC_ / 16, (B_ * SENC_) / 64), 128, 0, stream>>>(
        enc_bf, ENC_, Wk_bf, ENC_, pk, DEC_, ENC_);

    // ---- 800 sequential decoder steps ----
    for (int t = 0; t < T_; ++t) {
        bf16_t* XAc = (t & 1) ? XA1 : XA0;
        bf16_t* XAn = (t & 1) ? XA0 : XA1;
        bf16_t* XDc = (t & 1) ? XD1 : XD0;
        bf16_t* XDn = (t & 1) ? XD0 : XD1;

        set_mel_kernel<<<B_, MEL_, 0, stream>>>(XAc, tmel, t);

        // attention LSTM: gates from [mel|enc_last|pad|ah_{t-1}], writes
        // ah_t into XA_next (h region) and XD_cur[:,0:1024]
        lstm_gates_kernel<<<DEC_ / 16, 128, 0, stream>>>(
            XAc, KA_, KA_, WcatA, b_a, ac, XAn, KA_, KAH_, XDc, KD_, 0);

        // pq = ah_t @ Wq^T
        gemm_f32out<<<dim3(DEC_ / 16, 1), 128, 0, stream>>>(
            XDc, KD_, Wq_bf, DEC_, pq, DEC_, DEC_);

        attn_scores_kernel<<<dim3(B_, SENC_ / 8), 256, 0, stream>>>(pq, pk, We, scores);

        softmax_ctx_kernel<<<B_, 256, 0, stream>>>(scores, enc_bf, XDc);

        // decoder LSTM: gates from [ah_t|context_t|dh_{t-1}], writes dh_t
        // into XD_next (dh region)
        lstm_gates_kernel<<<DEC_ / 16, 128, 0, stream>>>(
            XDc, KD_, KD_, WcatD, b_d, dc, XDn, KD_, KDH_, (bf16_t*)nullptr, 0, 0);

        // mel_t = dh_t @ Wmel^T + bmel  -> out[:, t, :]
        mel_gemm_kernel<<<MEL_ / 16, 128, 0, stream>>>(
            XDn + KDH_, KD_, Wm_bf, bmel, out, t);
    }
}
